// Attention_7722351198977
// MI455X (gfx1250) — matrix operand order repaired and checked by reading, not compiled, not run on hardware
//
#include <hip/hip_runtime.h>
#include <hip/hip_bf16.h>

#define BB 8
#define TT 1024
#define DD 1024
#define HH 64

typedef __bf16 bf16;
typedef __attribute__((ext_vector_type(16))) __bf16 v16bf;
typedef __attribute__((ext_vector_type(8)))  __bf16 v8bf;
typedef __attribute__((ext_vector_type(8)))  float  v8f;
typedef __attribute__((ext_vector_type(4)))  float  v4f;

static __device__ __forceinline__ int lane_id() { return threadIdx.x & 31; }

static __device__ __forceinline__ v8f wmma_bf16(v16bf a, v16bf b, v8f c) {
  return __builtin_amdgcn_wmma_f32_16x16x32_bf16(false, a, false, b, (short)0, c,
                                                 false, false);
}

#define V8F_ZERO {0.f, 0.f, 0.f, 0.f, 0.f, 0.f, 0.f, 0.f}

// ---------------------------------------------------------------------------
// Fragment layouts per CDNA5 ISA 7.12.2 (wave32):
// A (16x32, 16-bit): lane L<16 -> row L,   K = {0..7, 16..23}
//                    lane L>=16 -> row L-16, K = {8..15, 24..31}
// B (32x16, 16-bit): lane L -> column L&15; lanes 0-15 hold K = {0..7, 16..23},
//                    lanes 16-31 hold K = {8..15, 24..31}.
// C/D (16x16 f32):   lane L -> column L&15, element i -> row i + 8*(L>>4)
// All B sources are pre-transposed in ws so B fragments are two 16B loads.
// ---------------------------------------------------------------------------

// A fragment from row-major fp32 source (converted to bf16), row stride ldr.
static __device__ __forceinline__ v16bf a_frag_from_f32(const float* src, int ldr) {
  const int L = lane_id();
  const int r = L & 15, h = L >> 4;
  const float* p = src + (size_t)r * ldr + h * 8;
  v16bf out;
#pragma unroll
  for (int c = 0; c < 2; ++c) {
    v4f x0 = *(const v4f*)(p + c * 16);
    v4f x1 = *(const v4f*)(p + c * 16 + 4);
#pragma unroll
    for (int j = 0; j < 4; ++j) {
      out[c * 8 + j]     = (bf16)x0[j];
      out[c * 8 + 4 + j] = (bf16)x1[j];
    }
  }
  return out;
}

// A fragment from row-major bf16 source, row stride ldr (elements).
static __device__ __forceinline__ v16bf a_frag_from_bf16(const bf16* src, int ldr) {
  const int L = lane_id();
  const int r = L & 15, h = L >> 4;
  const bf16* p = src + (size_t)r * ldr + h * 8;
  v8bf c0 = *(const v8bf*)(p);
  v8bf c1 = *(const v8bf*)(p + 16);
  v16bf out;
#pragma unroll
  for (int j = 0; j < 8; ++j) { out[j] = c0[j]; out[8 + j] = c1[j]; }
  return out;
}

// B fragment for C = A * S^T where S is row-major bf16 [n][K]: lane = row n of S.
// src points at (row 0, kbase); two contiguous 16B loads per lane.
static __device__ __forceinline__ v16bf bT_frag_from_bf16(const bf16* src, int ldr) {
  const int L = lane_id();
  const int n = L & 15, h = L >> 4;
  const bf16* p = src + (size_t)n * ldr + h * 8;
  v8bf c0 = *(const v8bf*)(p);
  v8bf c1 = *(const v8bf*)(p + 16);
  v16bf out;
#pragma unroll
  for (int j = 0; j < 8; ++j) { out[j] = c0[j]; out[8 + j] = c1[j]; }
  return out;
}

// Same but from fp32 source (converted): two runs of 8 floats per lane, at k = 8h and 16 + 8h.
static __device__ __forceinline__ v16bf bT_frag_from_f32(const float* src, int ldr) {
  const int L = lane_id();
  const int n = L & 15, h = L >> 4;
  const float* p = src + (size_t)n * ldr + h * 8;
  v16bf out;
#pragma unroll
  for (int c = 0; c < 4; ++c) {
    v4f x = *(const v4f*)(p + (c >> 1) * 16 + (c & 1) * 4);
#pragma unroll
    for (int j = 0; j < 4; ++j) out[c * 4 + j] = (bf16)x[j];
  }
  return out;
}

// Batched-q A fragment: rows = batch index (8 valid, 8 zero-padded), K = H.
static __device__ __forceinline__ v16bf qbatch_frag(const bf16* qb, int t, int kk) {
  const int L = lane_id();
  const int r = L & 15, h = L >> 4;
  v16bf a;
  if (r < BB) {
    const bf16* p = qb + ((size_t)r * TT + t) * HH + kk + h * 8;
    v8bf c0 = *(const v8bf*)(p);
    v8bf c1 = *(const v8bf*)(p + 16);
#pragma unroll
    for (int j = 0; j < 8; ++j) { a[j] = c0[j]; a[8 + j] = c1[j]; }
  } else {
#pragma unroll
    for (int j = 0; j < 16; ++j) a[j] = (bf16)0.0f;
  }
  return a;
}

static __device__ __forceinline__ void c_store_f32(float* dst, int ldr, v8f c) {
  const int L = lane_id();
  const int n = L & 15, h = L >> 4;
#pragma unroll
  for (int i = 0; i < 8; ++i) dst[(size_t)(h * 8 + i) * ldr + n] = c[i];
}

static __device__ __forceinline__ void c_store_bf16(bf16* dst, int ldr, v8f c) {
  const int L = lane_id();
  const int n = L & 15, h = L >> 4;
#pragma unroll
  for (int i = 0; i < 8; ++i) dst[(size_t)(h * 8 + i) * ldr + n] = (bf16)c[i];
}

// Transposed store: element (row M, col n) -> dst[n * ldc + M].
static __device__ __forceinline__ void c_store_bf16_T(bf16* dst, int ldc, v8f c) {
  const int L = lane_id();
  const int n = L & 15, h = L >> 4;
#pragma unroll
  for (int i = 0; i < 8; ++i) dst[(size_t)n * ldc + h * 8 + i] = (bf16)c[i];
}

// ---------------------------------------------------------------------------
// K0: transpose Wq/Wk/Wv (fp32 [D][H]) -> bf16 [H][D].  grid (256, 3), block 256.
// ---------------------------------------------------------------------------
__global__ void __launch_bounds__(256)
wt_kernel(const float* __restrict__ Wq, const float* __restrict__ Wk,
          const float* __restrict__ Wv, bf16* __restrict__ WtQ,
          bf16* __restrict__ WtK, bf16* __restrict__ WtV) {
  const int w = blockIdx.y;
  const float* W = (w == 0) ? Wq : ((w == 1) ? Wk : Wv);
  bf16* Wt = (w == 0) ? WtQ : ((w == 1) ? WtK : WtV);
  const int idx = blockIdx.x * 256 + threadIdx.x;  // 0 .. H*D-1
  const int h = idx >> 10, d = idx & (DD - 1);
  Wt[(size_t)h * DD + d] = (bf16)W[(size_t)d * HH + h];
}

// ---------------------------------------------------------------------------
// K1: fused q/k/v projections, x read once. grid (8192/16), block 32.
// One wave: 16 rows x 64 cols x 3 outputs, K-loop over D, 12 WMMAs/step.
// v is stored transposed (vT [B][H][T]) for the LDS-free AV GEMM.
// ---------------------------------------------------------------------------
__global__ void __launch_bounds__(32)
proj_kernel(const float* __restrict__ x, const bf16* __restrict__ WtQ,
            const bf16* __restrict__ WtK, const bf16* __restrict__ WtV,
            bf16* __restrict__ qb, bf16* __restrict__ kb, bf16* __restrict__ vT) {
  const int m0 = blockIdx.x * 16;               // flattened b*T + t row base
  const float* xrow = x + (size_t)m0 * DD;

  v8f zero = V8F_ZERO;
  v8f accq[4] = {zero, zero, zero, zero};
  v8f acck[4] = {zero, zero, zero, zero};
  v8f accv[4] = {zero, zero, zero, zero};

  for (int kk = 0; kk < DD; kk += 32) {
    v16bf a = a_frag_from_f32(xrow + kk, DD);
#pragma unroll
    for (int nb = 0; nb < 4; ++nb) {
      accq[nb] = wmma_bf16(a, bT_frag_from_bf16(WtQ + (size_t)(nb * 16) * DD + kk, DD), accq[nb]);
      acck[nb] = wmma_bf16(a, bT_frag_from_bf16(WtK + (size_t)(nb * 16) * DD + kk, DD), acck[nb]);
      accv[nb] = wmma_bf16(a, bT_frag_from_bf16(WtV + (size_t)(nb * 16) * DD + kk, DD), accv[nb]);
    }
  }

  const int b = m0 / TT, tb = m0 % TT;
#pragma unroll
  for (int nb = 0; nb < 4; ++nb) {
    c_store_bf16(qb + (size_t)m0 * HH + nb * 16, HH, accq[nb]);
    c_store_bf16(kb + (size_t)m0 * HH + nb * 16, HH, acck[nb]);
    c_store_bf16_T(vT + ((size_t)b * HH + nb * 16) * TT + tb, TT, accv[nb]);
  }
}

// ---------------------------------------------------------------------------
// K2a: scores = 8 * q . k^T, lower-triangle 16x16 tiles only.
// grid (64 s-tiles, 64 t-tiles, 8 batches), block 32.
// ---------------------------------------------------------------------------
__global__ void __launch_bounds__(32)
qk_kernel(const bf16* __restrict__ qb, const bf16* __restrict__ kb,
          float* __restrict__ scores) {
  const int st = blockIdx.x, tt = blockIdx.y, b = blockIdx.z;
  if (st > tt) return;  // block-uniform: EXEC stays all-ones for WMMA
  const int s0 = st * 16, t0 = tt * 16;
  const bf16* qrow = qb + ((size_t)b * TT + t0) * HH;
  const bf16* krow = kb + ((size_t)b * TT + s0) * HH;
  v8f c = V8F_ZERO;
#pragma unroll
  for (int kk = 0; kk < HH; kk += 32) {
    v16bf a = a_frag_from_bf16(qrow + kk, HH);
    v16bf bf = bT_frag_from_bf16(krow + kk, HH);
    c = wmma_bf16(a, bf, c);
  }
#pragma unroll
  for (int i = 0; i < 8; ++i) c[i] *= 8.0f;  // * sqrt(H)
  c_store_f32(scores + ((size_t)b * TT + t0) * TT + s0, TT, c);
}

// ---------------------------------------------------------------------------
// K2b: relpos bias, batched over b as the M dimension (HBM-critical kernel:
// streams the relpos lower triangle, ~128 MB). One wave handles 4 s-tiles so
// the q A-fragments are loaded once per 8 WMMAs; relpos lines for upcoming
// tiles are prefetched (global_prefetch_b8).
// grid (16 s-blocks of 64, 1024 t), block 32. Adds into scores where s <= t.
// ---------------------------------------------------------------------------
__global__ void __launch_bounds__(32)
bias_kernel(const bf16* __restrict__ qb, const float* __restrict__ relpos,
            float* __restrict__ scores) {
  const int t = blockIdx.y;
  const int sblk = blockIdx.x * 64;
  if (sblk > t) return;  // block-uniform
  const int L = lane_id();
  const int r = L & 15, h = L >> 4;

  // Prefetch relpos lines of the later s-tiles of this block (full 4KB/tile:
  // 32 lanes cover 16 rows x 2 x 128B).
  for (int st = 1; st < 4; ++st) {
    const int s0 = sblk + st * 16;
    if (s0 <= t)
      __builtin_prefetch(relpos + ((size_t)t * TT + s0 + r) * HH + h * 32, 0, 3);
  }

  v16bf a0 = qbatch_frag(qb, t, 0);
  v16bf a1 = qbatch_frag(qb, t, 32);

  for (int st = 0; st < 4; ++st) {
    const int s0 = sblk + st * 16;
    if (s0 > t) break;  // wave-uniform
    const float* rp = relpos + ((size_t)t * TT + s0) * HH;
    v8f c = V8F_ZERO;
    c = wmma_bf16(a0, bT_frag_from_f32(rp, HH), c);
    c = wmma_bf16(a1, bT_frag_from_f32(rp + 32, HH), c);
    // C rows = batch (valid on lanes 0-15 where M = 0..7), col = s0 + (L&15)
    if (h == 0) {
      const int s = s0 + r;
      if (s <= t) {
#pragma unroll
        for (int i = 0; i < 8; ++i)
          scores[((size_t)i * TT + t) * TT + s] += c[i];
      }
    }
  }
}

// ---------------------------------------------------------------------------
// K3: causal softmax over s in [0, t], write bf16 attn (zeros for s > t).
// grid 8192 rows, block 256.
// ---------------------------------------------------------------------------
__global__ void __launch_bounds__(256)
softmax_kernel(const float* __restrict__ scores, bf16* __restrict__ attnb) {
  const int m = blockIdx.x;  // b*T + t
  const int t = m & (TT - 1);
  const int len = t + 1;
  const float* row = scores + (size_t)m * TT;
  bf16* arow = attnb + (size_t)m * TT;
  __shared__ float red[256];
  const int tid = threadIdx.x;

  float mx = -3.4e38f;
  for (int s = tid; s < len; s += 256) mx = fmaxf(mx, row[s]);
  red[tid] = mx; __syncthreads();
  for (int off = 128; off > 0; off >>= 1) {
    if (tid < off) red[tid] = fmaxf(red[tid], red[tid + off]);
    __syncthreads();
  }
  mx = red[0]; __syncthreads();

  float sum = 0.f;
  for (int s = tid; s < len; s += 256) sum += __expf(row[s] - mx);
  red[tid] = sum; __syncthreads();
  for (int off = 128; off > 0; off >>= 1) {
    if (tid < off) red[tid] += red[tid + off];
    __syncthreads();
  }
  const float inv = 1.0f / red[0];

  for (int s = tid; s < TT; s += 256)
    arow[s] = (s < len) ? (bf16)(__expf(row[s] - mx) * inv) : (bf16)0.0f;
}

// ---------------------------------------------------------------------------
// K4: out = attn . v using transposed vT (LDS-free). grid (64, 8), block 32.
// One wave: 16 t-rows x 64 cols, K-loop truncated at the causal boundary.
// ---------------------------------------------------------------------------
__global__ void __launch_bounds__(32)
av_kernel(const bf16* __restrict__ attnb, const bf16* __restrict__ vT,
          float* __restrict__ out) {
  const int b = blockIdx.y;
  const int t0 = blockIdx.x * 16;
  const bf16* arow = attnb + ((size_t)b * TT + t0) * TT;
  const bf16* vTb = vT + (size_t)b * HH * TT;

  v8f zero = V8F_ZERO;
  v8f acc[4] = {zero, zero, zero, zero};

  const int kmax = t0 + 16;  // attn rows in this tile are zero past s = t0+15
  for (int kc = 0; kc < kmax; kc += 32) {  // uniform trip count per wave
    v16bf a = a_frag_from_bf16(arow + kc, TT);
#pragma unroll
    for (int nb = 0; nb < 4; ++nb) {
      v16bf bf = bT_frag_from_bf16(vTb + (size_t)(nb * 16) * TT + kc, TT);
      acc[nb] = wmma_bf16(a, bf, acc[nb]);
    }
  }
#pragma unroll
  for (int nb = 0; nb < 4; ++nb)
    c_store_f32(out + ((size_t)b * TT + t0) * HH + nb * 16, HH, acc[nb]);
}

// ---------------------------------------------------------------------------
extern "C" void kernel_launch(void* const* d_in, const int* in_sizes, int n_in,
                              void* d_out, int out_size, void* d_ws, size_t ws_size,
                              hipStream_t stream) {
  const float* x      = (const float*)d_in[0];
  const float* Wq     = (const float*)d_in[1];
  const float* Wk     = (const float*)d_in[2];
  const float* Wv     = (const float*)d_in[3];
  const float* relpos = (const float*)d_in[4];
  float* out = (float*)d_out;

  // Workspace layout:
  //   qb/kb   : bf16 [B][T][H]        (1 MiB each)
  //   vT      : bf16 [B][H][T]        (1 MiB)
  //   WtQ/K/V : bf16 [H][D]           (128 KiB each)
  //   scores  : f32  [B][T][T]        (32 MiB, L2-resident)
  //   attnb   : bf16 [B][T][T]        (16 MiB, L2-resident)
  char* ws = (char*)d_ws;
  bf16* qb  = (bf16*)(ws);
  bf16* kb  = (bf16*)(ws + (1u << 20));
  bf16* vT  = (bf16*)(ws + (2u << 20));
  bf16* WtQ = (bf16*)(ws + (3u << 20));
  bf16* WtK = (bf16*)(ws + (3u << 20) + (128u << 10));
  bf16* WtV = (bf16*)(ws + (3u << 20) + (256u << 10));
  float* scores = (float*)(ws + (4u << 20));
  bf16* attnb = (bf16*)(ws + (4u << 20) + (size_t)BB * TT * TT * sizeof(float));

  hipLaunchKernelGGL(wt_kernel, dim3((HH * DD) / 256, 3), dim3(256), 0, stream,
                     Wq, Wk, Wv, WtQ, WtK, WtV);
  hipLaunchKernelGGL(proj_kernel, dim3((BB * TT) / 16), dim3(32), 0, stream,
                     x, WtQ, WtK, WtV, qb, kb, vT);
  hipLaunchKernelGGL(qk_kernel, dim3(TT / 16, TT / 16, BB), dim3(32), 0, stream,
                     qb, kb, scores);
  hipLaunchKernelGGL(bias_kernel, dim3(TT / 64, TT), dim3(32), 0, stream,
                     qb, relpos, scores);
  hipLaunchKernelGGL(softmax_kernel, dim3(BB * TT), dim3(256), 0, stream,
                     scores, attnb);
  hipLaunchKernelGGL(av_kernel, dim3(TT / 16, BB), dim3(32), 0, stream,
                     attnb, vT, out);
}
